// SelfAttention_17660905521394
// MI455X (gfx1250) — matrix operand order repaired and checked by reading, not compiled, not run on hardware
//
#include <hip/hip_runtime.h>
#include <hip/hip_bf16.h>

// ---------------- types for WMMA fragments (wave32, gfx1250) ----------------
typedef __attribute__((ext_vector_type(16))) __bf16    v16bf;
typedef __attribute__((ext_vector_type(8)))  float     v8f;
typedef __attribute__((ext_vector_type(4)))  unsigned  u32x4;
typedef __attribute__((ext_vector_type(8)))  int       i32x8;
typedef __attribute__((ext_vector_type(4)))  int       i32x4;

union FragAB {            // 32 bytes = 8 VGPRs = 16 bf16 per lane
    u32x4 q[2];
    v16bf v;
};

__device__ __forceinline__ unsigned short f32_to_bf16(float f) {
    unsigned u = __float_as_uint(f);
    unsigned r = 0x7FFFu + ((u >> 16) & 1u);   // round-to-nearest-even
    return (unsigned short)((u + r) >> 16);
}

// ---------------- elementwise f32 -> bf16 convert ----------------
__global__ void cvt_f32_bf16(const float* __restrict__ in,
                             unsigned short* __restrict__ out, int n) {
    int i = blockIdx.x * blockDim.x + threadIdx.x;
    int stride = gridDim.x * blockDim.x;
    for (; i < n; i += stride) out[i] = f32_to_bf16(in[i]);
}

// ---------------- tiled bf16 GEMM with WMMA ----------------
// C[M,N] = A[M,K] * W[K,N] + bias[N]
// MODE 0: write f32 to outF[m*N+n]
// MODE 1: write bf16 to outB in [B,H,S,D]  (Q / K layout)
// MODE 2: write bf16 to outB in [B,H,D,S]  (V transposed for PV WMMA)
#define GE_BM 128
#define GE_BN 64
#define GE_BK 32

#define HAVE_TDM (__has_builtin(__builtin_amdgcn_tensor_load_to_lds) && \
                  __has_builtin(__builtin_amdgcn_s_wait_tensorcnt))

template<int MODE>
__global__ __launch_bounds__(256)
void gemm_bf16_wmma(const unsigned short* __restrict__ A,
                    const unsigned short* __restrict__ W,
                    const float* __restrict__ bias,
                    float* __restrict__ outF,
                    unsigned short* __restrict__ outB,
                    int M, int N, int K)
{
    __shared__ unsigned short As[GE_BM * GE_BK];   // [row][k]   128x32
    __shared__ unsigned short Bs[GE_BN * GE_BK];   // [n][k]     64x32 (transposed)

    const int tid    = threadIdx.x;
    const int lane   = tid & 31;
    const int wave   = tid >> 5;
    const int half16 = lane >> 4;
    const int l16    = lane & 15;
    const int kb     = half16 * 8;     // A-frag K split (K 0-7/8-15 then 16-23/24-31)
    const int h16o   = half16 * 16;    // unused; B-frags use kb, same K split as A

    const int nBlocks = N / GE_BN;
    const int bm = (int)blockIdx.x / nBlocks;
    const int bn = (int)blockIdx.x % nBlocks;
    const int row0 = bm * GE_BM;
    const int col0 = bn * GE_BN;

    const int wm = wave & 3;           // 4 waves along M (32 rows each)
    const int wn = wave >> 2;          // 2 waves along N (32 cols each)
    const int wrow = wm * 32;
    const int wcol = wn * 32;

    // staging assignments
    const int arow  = tid >> 2;        // 64 rows per pass, 4 threads/row * 8 elems
    const int akoff = (tid & 3) * 8;
    const int bkrow = tid >> 3;        // 32 k-rows, 8 threads/row * 8 elems
    const int bnoff = (tid & 7) * 8;

    v8f acc00 = {}, acc01 = {}, acc10 = {}, acc11 = {};

#if HAVE_TDM
    const unsigned ldsAoff = (unsigned)(uintptr_t)(void*)&As[0];  // LDS byte offset
#endif

    for (int k0 = 0; k0 < K; k0 += GE_BK) {
        // global loads for manual staging (B always; A only without TDM)
        u32x4 bv = *(const u32x4*)(W + (size_t)(k0 + bkrow) * N + col0 + bnoff);
#if !HAVE_TDM
        u32x4 av0 = *(const u32x4*)(A + (size_t)(row0 + arow) * K + k0 + akoff);
        u32x4 av1 = *(const u32x4*)(A + (size_t)(row0 + 64 + arow) * K + k0 + akoff);
#endif
        if (k0 + GE_BK < K)
            __builtin_prefetch((const void*)(W + (size_t)(k0 + GE_BK + bkrow) * N + col0 + bnoff), 0, 1);

        __syncthreads();               // previous iteration's consumers done

#if HAVE_TDM
        // ---- Tensor Data Mover: DMA the 128x32 bf16 A tile into LDS ----
        // D# per CDNA5 ISA (2D tensor, groups 2/3 zero)
        if (wave == 0) {
            unsigned long long ga =
                (unsigned long long)(uintptr_t)(A + (size_t)row0 * K + k0);
            u32x4 g0;
            g0[0] = 1u;                                     // count=1
            g0[1] = ldsAoff;                                // lds_addr (bytes)
            g0[2] = (unsigned)(ga & 0xFFFFFFFFu);           // global_addr lo
            g0[3] = (unsigned)((ga >> 32) & 0x1FFFFFFu)     // global_addr hi (56:32)
                    | 0x80000000u;                          // type=2 (image)
            i32x8 g1;
            g1[0] = (int)(1u << 16);                        // data_size=1 -> 2 bytes
            g1[1] = (int)((unsigned)(K & 0xFFFF) << 16);    // tensor_dim0[15:0]
            g1[2] = (int)(((unsigned)K >> 16) |             // tensor_dim0[31:16]
                          ((unsigned)(M & 0xFFFF) << 16));  // tensor_dim1[15:0]
            g1[3] = (int)(((unsigned)M >> 16) |             // tensor_dim1[31:16]
                          (32u << 16));                     // tile_dim0 = 32 (K)
            g1[4] = 128;                                    // tile_dim1=128, tile_dim2=0
            g1[5] = (int)K;                                 // tensor_dim0_stride lo
            g1[6] = 0;
            g1[7] = 0;
            i32x4 z4 = {0, 0, 0, 0};
#if defined(__clang_major__) && (__clang_major__ >= 23)
            i32x8 z8 = {0, 0, 0, 0, 0, 0, 0, 0};
            __builtin_amdgcn_tensor_load_to_lds(g0, g1, z4, z4, z8, 0);
#else
            __builtin_amdgcn_tensor_load_to_lds(g0, g1, z4, z4, 0);
#endif
            __builtin_amdgcn_s_wait_tensorcnt(0);
        }
#else
        *(u32x4*)(As + arow * GE_BK + akoff) = av0;
        *(u32x4*)(As + (64 + arow) * GE_BK + akoff) = av1;
#endif
        {   // transpose W tile into Bs[n][k]
            const unsigned short* bvs = (const unsigned short*)&bv;
            #pragma unroll
            for (int j = 0; j < 8; ++j)
                Bs[(bnoff + j) * GE_BK + bkrow] = bvs[j];
        }
        __syncthreads();

        FragAB a0, a1, b0, b1;
        a0.q[0] = *(const u32x4*)(As + (wrow + l16) * GE_BK + kb);
        a0.q[1] = *(const u32x4*)(As + (wrow + l16) * GE_BK + 16 + kb);
        a1.q[0] = *(const u32x4*)(As + (wrow + 16 + l16) * GE_BK + kb);
        a1.q[1] = *(const u32x4*)(As + (wrow + 16 + l16) * GE_BK + 16 + kb);
        b0.q[0] = *(const u32x4*)(Bs + (wcol + l16) * GE_BK + kb);
        b0.q[1] = *(const u32x4*)(Bs + (wcol + l16) * GE_BK + 16 + kb);
        b1.q[0] = *(const u32x4*)(Bs + (wcol + 16 + l16) * GE_BK + kb);
        b1.q[1] = *(const u32x4*)(Bs + (wcol + 16 + l16) * GE_BK + 16 + kb);

        acc00 = __builtin_amdgcn_wmma_f32_16x16x32_bf16(false, a0.v, false, b0.v,
                                                        (short)0, acc00, false, false);
        acc01 = __builtin_amdgcn_wmma_f32_16x16x32_bf16(false, a0.v, false, b1.v,
                                                        (short)0, acc01, false, false);
        acc10 = __builtin_amdgcn_wmma_f32_16x16x32_bf16(false, a1.v, false, b0.v,
                                                        (short)0, acc10, false, false);
        acc11 = __builtin_amdgcn_wmma_f32_16x16x32_bf16(false, a1.v, false, b1.v,
                                                        (short)0, acc11, false, false);
    }

    // epilogue: bias + scatter (straight-line, MODE is compile-time)
    v8f accs[2][2] = {{acc00, acc01}, {acc10, acc11}};
    #pragma unroll
    for (int rt = 0; rt < 2; ++rt) {
        #pragma unroll
        for (int ct = 0; ct < 2; ++ct) {
            const int colg = col0 + wcol + ct * 16 + l16;
            const float badd = bias[colg];
            #pragma unroll
            for (int i = 0; i < 8; ++i) {
                const int m = row0 + wrow + rt * 16 + half16 * 8 + i;
                const float val = accs[rt][ct][i] + badd;
                if (MODE == 0) {
                    outF[(size_t)m * N + colg] = val;
                } else {
                    const int bb = m >> 11;          // m / 2048
                    const int s  = m & 2047;
                    const int h  = colg >> 6;        // col / 64
                    const int d  = colg & 63;
                    size_t idx;
                    if (MODE == 1) idx = ((size_t)(bb * 16 + h) * 2048 + s) * 64 + d;
                    else           idx = ((size_t)(bb * 16 + h) * 64 + d) * 2048 + s;
                    outB[idx] = f32_to_bf16(val);
                }
            }
        }
    }
}

// ---------------- flash-style attention, one wave per 16-query tile ----------------
// Q,K: [B*H, S, D] bf16 ; Vt: [B*H, D, S] bf16 ; Og: [B, S, E] bf16
__global__ __launch_bounds__(256)
void attn_wmma(const unsigned short* __restrict__ Qg,
               const unsigned short* __restrict__ Kg,
               const unsigned short* __restrict__ Vt,
               unsigned short* __restrict__ Og)
{
    __shared__ unsigned short Plds[8 * 16 * 32];   // per-wave P tile (16x32)

    const int tid    = threadIdx.x;
    const int lane   = tid & 31;
    const int wave   = tid >> 5;
    const int half16 = lane >> 4;
    const int l16    = lane & 15;
    const int kb     = half16 * 8;
    const int h16o   = half16 * 16;

    const int task = blockIdx.x * 8 + wave;   // 8192 tasks total
    const int qt = task & 127;                // 128 q-tiles per head
    const int bh = task >> 7;                 // b*16+h
    const int h  = bh & 15;
    const int b  = bh >> 4;

    const unsigned short* Qp = Qg + ((size_t)bh * 2048 + qt * 16) * 64;
    const unsigned short* Kp = Kg + (size_t)bh * 2048 * 64;
    const unsigned short* Vp = Vt + (size_t)bh * 64 * 2048;
    unsigned short* Pw = Plds + wave * (16 * 32);

    // Q A-fragments (d = 0..31 and 32..63), loaded once
    const unsigned short* qrow = Qp + (size_t)l16 * 64;
    FragAB qa0, qa1;
    qa0.q[0] = *(const u32x4*)(qrow + kb);
    qa0.q[1] = *(const u32x4*)(qrow + 16 + kb);
    qa1.q[0] = *(const u32x4*)(qrow + 32 + kb);
    qa1.q[1] = *(const u32x4*)(qrow + 48 + kb);

    float mrow[8], lrow[8];
    #pragma unroll
    for (int i = 0; i < 8; ++i) { mrow[i] = -3.0e38f; lrow[i] = 0.0f; }
    v8f O[4] = {};                         // 16x64 f32 accumulator

    const float scale = 0.125f;            // 1/sqrt(64)

    for (int kc = 0; kc < 2048; kc += 32) {
        // ---- scores for 32 keys: two 16x16 tiles, each QK^T over D=64 ----
        v8f s[2];
        #pragma unroll
        for (int khf = 0; khf < 2; ++khf) {
            const unsigned short* krow = Kp + (size_t)(kc + khf * 16 + l16) * 64;
            FragAB b0, b1;
            b0.q[0] = *(const u32x4*)(krow + kb);
            b0.q[1] = *(const u32x4*)(krow + 16 + kb);
            b1.q[0] = *(const u32x4*)(krow + 32 + kb);
            b1.q[1] = *(const u32x4*)(krow + 48 + kb);
            v8f c = {};
            c = __builtin_amdgcn_wmma_f32_16x16x32_bf16(false, qa0.v, false, b0.v,
                                                        (short)0, c, false, false);
            c = __builtin_amdgcn_wmma_f32_16x16x32_bf16(false, qa1.v, false, b1.v,
                                                        (short)0, c, false, false);
            s[khf] = c * scale;
        }

        // ---- online softmax (f32), butterfly reductions over the 16 columns ----
        float alpha[8];
        #pragma unroll
        for (int i = 0; i < 8; ++i) {
            float cm = fmaxf(s[0][i], s[1][i]);
            #pragma unroll
            for (int off = 1; off < 16; off <<= 1)
                cm = fmaxf(cm, __shfl_xor(cm, off, 32));
            const float mn = fmaxf(mrow[i], cm);
            alpha[i] = __expf(mrow[i] - mn);
            mrow[i]  = mn;
            const float p0 = __expf(s[0][i] - mn);
            const float p1 = __expf(s[1][i] - mn);
            s[0][i] = p0; s[1][i] = p1;
            float rs = p0 + p1;
            #pragma unroll
            for (int off = 1; off < 16; off <<= 1)
                rs += __shfl_xor(rs, off, 32);
            lrow[i] = lrow[i] * alpha[i] + rs;
        }

        // ---- re-layout P (C layout -> A layout) via per-wave LDS ----
        #pragma unroll
        for (int i = 0; i < 8; ++i) {
            const int r = half16 * 8 + i;
            Pw[r * 32 + l16]      = f32_to_bf16(s[0][i]);
            Pw[r * 32 + 16 + l16] = f32_to_bf16(s[1][i]);
        }
        asm volatile("s_wait_dscnt 0" ::: "memory");
        FragAB pa;
        pa.q[0] = *(const u32x4*)(Pw + l16 * 32 + kb);
        pa.q[1] = *(const u32x4*)(Pw + l16 * 32 + 16 + kb);

        // ---- O = O*alpha + P @ V (4 column tiles of 16) ----
        #pragma unroll
        for (int nt = 0; nt < 4; ++nt) {
            #pragma unroll
            for (int i = 0; i < 8; ++i) O[nt][i] *= alpha[i];
            const unsigned short* vrow = Vp + (size_t)(nt * 16 + l16) * 2048 + kc;
            FragAB vb;
            vb.q[0] = *(const u32x4*)(vrow + kb);
            vb.q[1] = *(const u32x4*)(vrow + 16 + kb);
            O[nt] = __builtin_amdgcn_wmma_f32_16x16x32_bf16(false, pa.v, false, vb.v,
                                                            (short)0, O[nt], false, false);
        }
    }

    // ---- normalize and write combined heads: [B,S,E] bf16 ----
    #pragma unroll
    for (int nt = 0; nt < 4; ++nt) {
        #pragma unroll
        for (int i = 0; i < 8; ++i) {
            const int r    = half16 * 8 + i;
            const int srow = qt * 16 + r;
            const int col  = h * 64 + nt * 16 + l16;
            const float val = O[nt][i] / lrow[i];
            Og[((size_t)b * 2048 + srow) * 1024 + col] = f32_to_bf16(val);
        }
    }
}

// ---------------- host launch ----------------
extern "C" void kernel_launch(void* const* d_in, const int* in_sizes, int n_in,
                              void* d_out, int out_size, void* d_ws, size_t ws_size,
                              hipStream_t stream) {
    const float* X  = (const float*)d_in[0];
    const float* Wq = (const float*)d_in[1];
    const float* bq = (const float*)d_in[2];
    const float* Wk = (const float*)d_in[3];
    const float* bk = (const float*)d_in[4];
    const float* Wv = (const float*)d_in[5];
    const float* bv = (const float*)d_in[6];
    const float* Wo = (const float*)d_in[7];
    const float* bo = (const float*)d_in[8];
    float* out = (float*)d_out;

    const int E = 1024, M = 4 * 2048;          // 8192 rows
    const size_t ME = (size_t)M * E;
    const size_t EE = (size_t)E * E;

    unsigned short* p   = (unsigned short*)d_ws;
    unsigned short* Xb  = p; p += ME;
    unsigned short* Wqb = p; p += EE;
    unsigned short* Wkb = p; p += EE;
    unsigned short* Wvb = p; p += EE;
    unsigned short* Wob = p; p += EE;
    unsigned short* Qb  = p; p += ME;
    unsigned short* Kb  = p; p += ME;
    unsigned short* Vtb = p; p += ME;
    unsigned short* Ab  = p; p += ME;

    cvt_f32_bf16<<<4096, 256, 0, stream>>>(X,  Xb,  (int)ME);
    cvt_f32_bf16<<<1024, 256, 0, stream>>>(Wq, Wqb, (int)EE);
    cvt_f32_bf16<<<1024, 256, 0, stream>>>(Wk, Wkb, (int)EE);
    cvt_f32_bf16<<<1024, 256, 0, stream>>>(Wv, Wvb, (int)EE);
    cvt_f32_bf16<<<1024, 256, 0, stream>>>(Wo, Wob, (int)EE);

    const int gemmBlocks = (M / GE_BM) * (E / GE_BN);   // 64 * 16 = 1024
    gemm_bf16_wmma<1><<<gemmBlocks, 256, 0, stream>>>(Xb, Wqb, bq, nullptr, Qb,  M, E, E);
    gemm_bf16_wmma<1><<<gemmBlocks, 256, 0, stream>>>(Xb, Wkb, bk, nullptr, Kb,  M, E, E);
    gemm_bf16_wmma<2><<<gemmBlocks, 256, 0, stream>>>(Xb, Wvb, bv, nullptr, Vtb, M, E, E);

    attn_wmma<<<1024, 256, 0, stream>>>(Qb, Kb, Vtb, Ab);

    gemm_bf16_wmma<0><<<gemmBlocks, 256, 0, stream>>>(Ab, Wob, bo, out, nullptr, M, E, E);
}
